// ImprovedComplexMultiHeadAttention_58059367907724
// MI455X (gfx1250) — matrix operand order repaired and checked by reading, not compiled, not run on hardware
//
#include <hip/hip_runtime.h>
#include <hip/hip_bf16.h>
#include <math.h>

// Problem constants (match the reference)
#define B_  4
#define D_  512
#define S_  2048
#define H_  8
#define DK_ 64

typedef __bf16 bf16;
typedef bf16  v16bf __attribute__((ext_vector_type(16)));
typedef bf16  v8bf  __attribute__((ext_vector_type(8)));
typedef float v8f   __attribute__((ext_vector_type(8)));

// ---------------------------------------------------------------------------
// WMMA helpers (CDNA5 wave32, 16x16x32 bf16 -> f32)
// ---------------------------------------------------------------------------
__device__ __forceinline__ v8f wmma_bf16(v16bf a, v16bf b, v8f c) {
  // 8 args: (neg_a, A, neg_b, B, c_mod, C, reuse_a, reuse_b)
  return __builtin_amdgcn_wmma_f32_16x16x32_bf16(false, a, false, b, (short)0, c,
                                                 false, false);
}

// A-matrix fragment (16x32, M x K), memory layout [m][k] with k contiguous.
__device__ __forceinline__ v16bf load_frag_a(const bf16* p, int ld) {
  const int lane = (int)(threadIdx.x & 31);
  const bf16* q = p + (size_t)(lane & 15) * ld + ((lane >> 4) << 4);
  v8bf lo = *(const v8bf*)(q);        // K = khalf .. khalf+7
  v8bf hi = *(const v8bf*)(q + 8);    // K = khalf+8 .. khalf+15
  return __builtin_shufflevector(lo, hi, 0, 1, 2, 3, 4, 5, 6, 7,
                                 8, 9, 10, 11, 12, 13, 14, 15);
}

// B-matrix fragment (32x16, K x N), memory layout [n][k] with k contiguous.
__device__ __forceinline__ v16bf load_frag_b(const bf16* p, int ld) {
  const int lane = (int)(threadIdx.x & 31);
  const bf16* q = p + (size_t)(lane & 15) * ld + ((lane >> 4) << 4);
  return *(const v16bf*)q;
}

// Row reductions over a 16-lane group (C/D rows span 16 lanes on wave32).
__device__ __forceinline__ float rowmax16(float v) {
  v = fmaxf(v, __shfl_xor(v, 1, 16));
  v = fmaxf(v, __shfl_xor(v, 2, 16));
  v = fmaxf(v, __shfl_xor(v, 4, 16));
  v = fmaxf(v, __shfl_xor(v, 8, 16));
  return v;
}
__device__ __forceinline__ float rowsum16(float v) {
  v += __shfl_xor(v, 1, 16);
  v += __shfl_xor(v, 2, 16);
  v += __shfl_xor(v, 4, 16);
  v += __shfl_xor(v, 8, 16);
  return v;
}

// ---------------------------------------------------------------------------
// Kernel 1: fp32 -> bf16 cast (weights)
// ---------------------------------------------------------------------------
__global__ __launch_bounds__(256) void cvt_f32_bf16_kernel(
    const float* __restrict__ src, bf16* __restrict__ dst, int n) {
  int i = blockIdx.x * blockDim.x + threadIdx.x;
  if (i < n) dst[i] = (bf16)src[i];
}

// ---------------------------------------------------------------------------
// Kernel 2: x (B,D,2,S) fp32 -> xT (B,3,S,D) bf16, planes = {xr, xi, -xi}
// ---------------------------------------------------------------------------
__global__ __launch_bounds__(256) void transpose_x_kernel(
    const float* __restrict__ x, bf16* __restrict__ xT) {
  __shared__ float tile[32][33];
  const int bri = blockIdx.z;  // b*2 + ri
  const int b = bri >> 1, ri = bri & 1;
  const int s0 = blockIdx.x * 32, c0 = blockIdx.y * 32;
  const int tx = threadIdx.x, ty = threadIdx.y;  // block (32, 8)
#pragma unroll
  for (int yy = 0; yy < 4; ++yy) {
    int c = c0 + ty + yy * 8;
    int s = s0 + tx;
    tile[ty + yy * 8][tx] = x[(((size_t)b * D_ + c) * 2 + ri) * S_ + s];
  }
  __syncthreads();
#pragma unroll
  for (int yy = 0; yy < 4; ++yy) {
    int srow = s0 + ty + yy * 8;
    int ccol = c0 + tx;
    float v = tile[tx][ty + yy * 8];
    if (ri == 0) {
      xT[(((size_t)b * 3 + 0) * S_ + srow) * D_ + ccol] = (bf16)v;
    } else {
      xT[(((size_t)b * 3 + 1) * S_ + srow) * D_ + ccol] = (bf16)v;
      xT[(((size_t)b * 3 + 2) * S_ + srow) * D_ + ccol] = (bf16)(-v);
    }
  }
}

// ---------------------------------------------------------------------------
// Kernel 3: complex projection GEMM (WMMA bf16)
//   accR = Wr*xr + Wi*(-xi)   -> o_r = accR + (br-bi)
//   accI = Wr*xi + Wi*xr      -> o_i = accI + (br+bi)
// 8 waves (4 m-groups x 2 n-groups), wave tile 32x32. Only 8 v8f accumulators.
// mode 0: magnitude*mag_scale transposed -> outM[b][s][o]   (Q/K)
// mode 1: bf16 planes -> outRb/outIb [b][o][s]               (V)
// mode 2: fp32 interleaved -> outF (B,D,2,S)                 (O-proj)
// ---------------------------------------------------------------------------
__global__ __launch_bounds__(256, 1) void cproj_kernel(
    const bf16* __restrict__ Wr, const bf16* __restrict__ Wi,
    const float* __restrict__ br, const float* __restrict__ bi,
    const bf16* __restrict__ xT, int mode, float mag_scale,
    bf16* __restrict__ outM, bf16* __restrict__ outRb,
    bf16* __restrict__ outIb, float* __restrict__ outF) {
  const int b = blockIdx.z;
  const int lane = (int)(threadIdx.x & 31);
  const int wave = (int)(threadIdx.x >> 5);
  const int wm = wave >> 1, wn = wave & 1;
  const int o0 = blockIdx.y * 128 + wm * 32;
  const int s0 = blockIdx.x * 64 + wn * 32;

  v8f accR[2][2] = {}, accI[2][2] = {};

  const bf16* xr = xT + (((size_t)b * 3 + 0) * S_ + s0) * D_;
  const bf16* xi = xT + (((size_t)b * 3 + 1) * S_ + s0) * D_;
  const bf16* xn = xT + (((size_t)b * 3 + 2) * S_ + s0) * D_;

#pragma unroll 1
  for (int c0 = 0; c0 < D_; c0 += 32) {
    if (c0 + 32 < D_) {  // prefetch next weight K-slab (global_prefetch_b8)
      __builtin_prefetch(Wr + (size_t)o0 * D_ + c0 + 32, 0, 1);
      __builtin_prefetch(Wi + (size_t)o0 * D_ + c0 + 32, 0, 1);
    }
    v16bf ar[2], ai[2];
#pragma unroll
    for (int m = 0; m < 2; ++m) {
      ar[m] = load_frag_a(Wr + (size_t)(o0 + m * 16) * D_ + c0, D_);
      ai[m] = load_frag_a(Wi + (size_t)(o0 + m * 16) * D_ + c0, D_);
    }
#pragma unroll
    for (int n = 0; n < 2; ++n) {
      v16bf bre = load_frag_b(xr + (size_t)(n * 16) * D_ + c0, D_);
      v16bf bim = load_frag_b(xi + (size_t)(n * 16) * D_ + c0, D_);
      v16bf bin = load_frag_b(xn + (size_t)(n * 16) * D_ + c0, D_);
#pragma unroll
      for (int m = 0; m < 2; ++m) {
        accR[m][n] = wmma_bf16(ar[m], bre, accR[m][n]);
        accR[m][n] = wmma_bf16(ai[m], bin, accR[m][n]);
        accI[m][n] = wmma_bf16(ar[m], bim, accI[m][n]);
        accI[m][n] = wmma_bf16(ai[m], bre, accI[m][n]);
      }
    }
  }

  const int rowg = (lane >> 4) * 8;
  const int colg = lane & 15;
#pragma unroll
  for (int m = 0; m < 2; ++m) {
    float bdiff[8], bsum[8];
#pragma unroll
    for (int r = 0; r < 8; ++r) {
      int row = o0 + m * 16 + rowg + r;
      float vr = br[row], vi = bi[row];
      bdiff[r] = vr - vi;
      bsum[r] = vr + vi;
    }
#pragma unroll
    for (int n = 0; n < 2; ++n) {
#pragma unroll
      for (int r = 0; r < 8; ++r) {
        int row = o0 + m * 16 + rowg + r;
        int s = s0 + n * 16 + colg;
        float o_r = accR[m][n][r] + bdiff[r];
        float o_i = accI[m][n][r] + bsum[r];
        if (mode == 0) {
          float mag = sqrtf(o_r * o_r + o_i * o_i + 1e-8f) * mag_scale;
          outM[((size_t)b * S_ + s) * D_ + row] = (bf16)mag;
        } else if (mode == 1) {
          outRb[((size_t)b * D_ + row) * S_ + s] = (bf16)o_r;
          outIb[((size_t)b * D_ + row) * S_ + s] = (bf16)o_i;
        } else {
          outF[(((size_t)b * D_ + row) * 2 + 0) * S_ + s] = o_r;
          outF[(((size_t)b * D_ + row) * 2 + 1) * S_ + s] = o_i;
        }
      }
    }
  }
}

// ---------------------------------------------------------------------------
// Kernel 4: flash attention on magnitudes.
// 8 independent waves; wave owns 16 s-rows (small accumulator footprint).
// Output planes {out_r, out_i, -out_i} so the O-projection can reuse cproj.
// ---------------------------------------------------------------------------
__global__ __launch_bounds__(256, 1) void flash_attn_kernel(
    const bf16* __restrict__ QmT, const bf16* __restrict__ KmT,
    const bf16* __restrict__ Vrb, const bf16* __restrict__ Vib,
    bf16* __restrict__ attT) {
  __shared__ bf16 pLds[8][16 * 32];
  const int b = blockIdx.z, h = blockIdx.y;
  const int wave = (int)(threadIdx.x >> 5);
  const int lane = (int)(threadIdx.x & 31);
  const int s0 = blockIdx.x * 128 + wave * 16;
  const int c0 = h * DK_;
  const int rowg = (lane >> 4) * 8;
  const int colg = lane & 15;

  // Q fragments (only persistent A operands)
  v16bf qa[2];
#pragma unroll
  for (int k = 0; k < 2; ++k)
    qa[k] = load_frag_a(QmT + ((size_t)b * S_ + s0) * D_ + c0 + k * 32, D_);

  v8f oR[4] = {}, oI[4] = {};
  float mstat[8], lstat[8];
#pragma unroll
  for (int r = 0; r < 8; ++r) {
    mstat[r] = -1e30f;
    lstat[r] = 0.0f;
  }

  bf16* pw = pLds[wave];

#pragma unroll 1
  for (int t0 = 0; t0 < S_; t0 += 32) {
    // ---- scores = Qm^T Km (16 s-rows x 32 t-cols), K-dim = DK = 64 ----
    v8f sc[2] = {};
#pragma unroll
    for (int n = 0; n < 2; ++n) {
      v16bf kb0 = load_frag_b(
          KmT + ((size_t)b * S_ + t0 + n * 16) * D_ + c0, D_);
      v16bf kb1 = load_frag_b(
          KmT + ((size_t)b * S_ + t0 + n * 16) * D_ + c0 + 32, D_);
      sc[n] = wmma_bf16(qa[0], kb0, sc[n]);
      sc[n] = wmma_bf16(qa[1], kb1, sc[n]);
    }

    // ---- online softmax update ----
#pragma unroll
    for (int r = 0; r < 8; ++r) {
      float rv = rowmax16(fmaxf(sc[0][r], sc[1][r]));
      float mnew = fmaxf(mstat[r], rv);
      float corr = __expf(mstat[r] - mnew);
      float p0 = __expf(sc[0][r] - mnew);
      float p1 = __expf(sc[1][r] - mnew);
      sc[0][r] = p0;
      sc[1][r] = p1;
      lstat[r] = lstat[r] * corr + rowsum16(p0 + p1);
      mstat[r] = mnew;
#pragma unroll
      for (int nd = 0; nd < 4; ++nd) {
        oR[nd][r] *= corr;
        oI[nd][r] *= corr;
      }
    }

    // ---- reshape P: C-layout -> A-layout via per-wave LDS tile ----
#pragma unroll
    for (int n = 0; n < 2; ++n)
#pragma unroll
      for (int r = 0; r < 8; ++r)
        pw[(rowg + r) * 32 + n * 16 + colg] = (bf16)sc[n][r];
    __syncthreads();

    v16bf ap = load_frag_a(pw, 32);
#pragma unroll
    for (int nd = 0; nd < 4; ++nd) {
      v16bf vr =
          load_frag_b(Vrb + ((size_t)b * D_ + c0 + nd * 16) * S_ + t0, S_);
      oR[nd] = wmma_bf16(ap, vr, oR[nd]);
      v16bf vi =
          load_frag_b(Vib + ((size_t)b * D_ + c0 + nd * 16) * S_ + t0, S_);
      oI[nd] = wmma_bf16(ap, vi, oI[nd]);
    }
    __syncthreads();
  }

  // ---- normalize, store planes {r, i, -i}: attT[b][pl][s][c] bf16 ----
#pragma unroll
  for (int nd = 0; nd < 4; ++nd)
#pragma unroll
    for (int r = 0; r < 8; ++r) {
      float inv = 1.0f / lstat[r];
      int srow = s0 + rowg + r;
      int col = c0 + nd * 16 + colg;
      float vr = oR[nd][r] * inv;
      float vi = oI[nd][r] * inv;
      attT[(((size_t)b * 3 + 0) * S_ + srow) * D_ + col] = (bf16)vr;
      attT[(((size_t)b * 3 + 1) * S_ + srow) * D_ + col] = (bf16)vi;
      attT[(((size_t)b * 3 + 2) * S_ + srow) * D_ + col] = (bf16)(-vi);
    }
}

// ---------------------------------------------------------------------------
// Host-side orchestration
// ---------------------------------------------------------------------------
extern "C" void kernel_launch(void* const* d_in, const int* in_sizes, int n_in,
                              void* d_out, int out_size, void* d_ws,
                              size_t ws_size, hipStream_t stream) {
  (void)in_sizes; (void)n_in; (void)out_size; (void)ws_size;

  const float* x = (const float*)d_in[0];
  const float* Wr[4] = {(const float*)d_in[1], (const float*)d_in[5],
                        (const float*)d_in[9], (const float*)d_in[13]};
  const float* Wi[4] = {(const float*)d_in[2], (const float*)d_in[6],
                        (const float*)d_in[10], (const float*)d_in[14]};
  const float* br[4] = {(const float*)d_in[3], (const float*)d_in[7],
                        (const float*)d_in[11], (const float*)d_in[15]};
  const float* bi[4] = {(const float*)d_in[4], (const float*)d_in[8],
                        (const float*)d_in[12], (const float*)d_in[16]};

  char* p = (char*)d_ws;
  bf16* xT = (bf16*)p;
  p += (size_t)B_ * 3 * S_ * D_ * sizeof(bf16);  // 24 MB (planes r, i, -i)
  bf16* Wbf[8];
  for (int i = 0; i < 8; ++i) {
    Wbf[i] = (bf16*)p;
    p += (size_t)D_ * D_ * sizeof(bf16);  // 0.5 MB each
  }
  bf16* QmT = (bf16*)p; p += (size_t)B_ * S_ * D_ * sizeof(bf16);       // 8 MB
  bf16* KmT = (bf16*)p; p += (size_t)B_ * S_ * D_ * sizeof(bf16);       // 8 MB
  bf16* Vrb = (bf16*)p; p += (size_t)B_ * D_ * S_ * sizeof(bf16);       // 8 MB
  bf16* Vib = (bf16*)p; p += (size_t)B_ * D_ * S_ * sizeof(bf16);       // 8 MB
  bf16* attT = (bf16*)p; p += (size_t)B_ * 3 * S_ * D_ * sizeof(bf16);  // 24 MB

  // 1) weights fp32 -> bf16
  const int wn = D_ * D_;
  for (int j = 0; j < 4; ++j) {
    cvt_f32_bf16_kernel<<<(wn + 255) / 256, 256, 0, stream>>>(Wr[j], Wbf[2 * j], wn);
    cvt_f32_bf16_kernel<<<(wn + 255) / 256, 256, 0, stream>>>(Wi[j], Wbf[2 * j + 1], wn);
  }

  // 2) transpose x -> xT[b][{r,i,-i}][s][c] bf16
  transpose_x_kernel<<<dim3(S_ / 32, D_ / 32, B_ * 2), dim3(32, 8), 0, stream>>>(x, xT);

  dim3 pg(S_ / 64, D_ / 128, B_);
  // 3) Q projection -> |Q| * (1/sqrt(DK)), transposed
  cproj_kernel<<<pg, 256, 0, stream>>>(Wbf[0], Wbf[1], br[0], bi[0], xT, 0,
                                       0.125f, QmT, nullptr, nullptr, nullptr);
  // 4) K projection -> |K|, transposed
  cproj_kernel<<<pg, 256, 0, stream>>>(Wbf[2], Wbf[3], br[1], bi[1], xT, 0,
                                       1.0f, KmT, nullptr, nullptr, nullptr);
  // 5) V projection -> bf16 planes [b][c][s]
  cproj_kernel<<<pg, 256, 0, stream>>>(Wbf[4], Wbf[5], br[2], bi[2], xT, 1,
                                       1.0f, nullptr, Vrb, Vib, nullptr);

  // 6) flash attention -> attT[b][{r,i,-i}][s][c] bf16
  flash_attn_kernel<<<dim3(S_ / 128, H_, B_), 256, 0, stream>>>(QmT, KmT, Vrb,
                                                                Vib, attT);

  // 7) output projection -> d_out (B, D, 2, S) fp32
  cproj_kernel<<<pg, 256, 0, stream>>>(Wbf[6], Wbf[7], br[3], bi[3], attT, 2,
                                       1.0f, nullptr, nullptr, nullptr,
                                       (float*)d_out);
}
